// MHA_Fused_90572270338234
// MI455X (gfx1250) — matrix operand order repaired and checked by reading, not compiled, not run on hardware
//
#include <hip/hip_runtime.h>
#include <hip/hip_bf16.h>

// ---------------------------------------------------------------------------
// MHA with sp2norm attention, fused for MI455X (gfx1250, wave32, WMMA bf16)
// B=2, T=2048, C=1024, H=16, hs=64
// ---------------------------------------------------------------------------

typedef __bf16 bf16;
typedef __attribute__((ext_vector_type(16))) __bf16 v16bf;
typedef __attribute__((ext_vector_type(8)))  __bf16 bf16x8;
typedef __attribute__((ext_vector_type(8)))  float   v8f;
typedef __attribute__((ext_vector_type(4)))  int     int4v;

#define Tdim 2048
#define Cdim 1024
#define Hn   16
#define HS   64
#define Bn   2
#define Mrows (Bn * Tdim)      // 4096
#define Nqkv  (3 * Cdim)       // 3072

union ABFrag { v16bf v; bf16x8 h[2]; };

// A fragment: 16x32 tile of row-major A (stride lda), tile origin (row0,k0).
// lane 0-15 -> M=lane, elems 0..7 = K kb..kb+7, elems 8..15 = K 16+kb..
static __device__ __forceinline__ v16bf load_a_frag(const bf16* A, int lda,
                                                    int row0, int k0, int lane) {
  int m  = lane & 15;
  int kb = (lane >> 4) << 3;               // 0 or 8
  const bf16* p = A + (size_t)(row0 + m) * (size_t)lda + (size_t)(k0 + kb);
  ABFrag f;
  f.h[0] = *(const bf16x8*)(p);
  f.h[1] = *(const bf16x8*)(p + 16);
  return f.v;
}

// B fragment: 32x16 tile of row-major B[k][n] (stride ldb), origin (k0,n0).
// lane -> N = n0+lane%16, elems 0..7 = K kb..kb+7, elems 8..15 = K 16+kb.. (kb = 8*(lane/16)).
// Works for both global and LDS (generic pointer; addrspace inferred).
static __device__ __forceinline__ v16bf load_b_frag(const bf16* B, int ldb,
                                                    int k0, int n0, int lane) {
  const bf16* p = B + (size_t)(k0 + ((lane >> 4) << 3)) * (size_t)ldb + (size_t)(n0 + (lane & 15));
  v16bf f;
  for (int s = 0; s < 8; ++s) {
    f[s]     = p[(size_t)s * ldb];
    f[s + 8] = p[(size_t)(16 + s) * ldb];
  }
  return f;
}

static __device__ __forceinline__ v8f wmma_bf16(v16bf a, v16bf b, v8f c) {
  return __builtin_amdgcn_wmma_f32_16x16x32_bf16(false, a, false, b,
                                                 (short)0, c, false, false);
}

static __device__ __forceinline__ float softplus_f(float x) {
  if (x > 20.0f) return x;                 // avoid exp overflow
  return __logf(1.0f + __expf(x));
}

// ---- CDNA5 async global->LDS path (ASYNCcnt), with portable fallback -------
#if defined(__has_builtin)
#if __has_builtin(__builtin_amdgcn_global_load_async_to_lds_b128) && \
    __has_builtin(__builtin_amdgcn_s_wait_asynccnt)
#define HAVE_ASYNC_LDS 1
#endif
#endif

// copy 16 bytes global -> LDS (per-lane addresses)
static __device__ __forceinline__ void copy16(bf16* dst_lds, const bf16* src_g) {
#ifdef HAVE_ASYNC_LDS
  __builtin_amdgcn_global_load_async_to_lds_b128(
      (__attribute__((address_space(1))) int4v*)(unsigned long long)src_g,
      (__attribute__((address_space(3))) int4v*)(unsigned int)(unsigned long long)dst_lds,
      0, 0);
#else
  *(bf16x8*)dst_lds = *(const bf16x8*)src_g;
#endif
}

static __device__ __forceinline__ void wait_async_pending() {
#ifdef HAVE_ASYNC_LDS
  __builtin_amdgcn_s_wait_asynccnt(2);     // previous chunk done, next in flight
#endif
}
static __device__ __forceinline__ void wait_async_all() {
#ifdef HAVE_ASYNC_LDS
  __builtin_amdgcn_s_wait_asynccnt(0);
#endif
}

// ---------------------------------------------------------------------------
// fp32 -> bf16 elementwise convert
__global__ __launch_bounds__(256) void cvt_bf16_kernel(const float* __restrict__ in,
                                                       bf16* __restrict__ out, int n) {
  int i = blockIdx.x * 256 + threadIdx.x;
  if (i < n) out[i] = (bf16)in[i];
}

// fp32 [rows][cols] -> bf16 transposed [cols][rows]
__global__ __launch_bounds__(256) void cvt_t_bf16_kernel(const float* __restrict__ in,
                                                         bf16* __restrict__ out,
                                                         int rows, int cols) {
  int i = blockIdx.x * 256 + threadIdx.x;
  if (i < rows * cols) {
    int r = i / cols, c = i - r * cols;
    out[(size_t)c * rows + r] = (bf16)in[i];
  }
}

// ---------------------------------------------------------------------------
// QKV GEMM: qkv[m][n] = sum_k Xb[m][k] * Wt[k][n], scatter to Q / Kt / V (bf16)
// One wave computes a 32x32 tile. Grid: (Mrows/32)*(Nqkv/32) waves.
__global__ __launch_bounds__(256) void qkv_gemm_kernel(const bf16* __restrict__ Xb,
                                                       const bf16* __restrict__ Wt,
                                                       bf16* __restrict__ Q,
                                                       bf16* __restrict__ Kt,
                                                       bf16* __restrict__ V) {
  int wave = blockIdx.x * (blockDim.x >> 5) + (threadIdx.x >> 5);
  int lane = threadIdx.x & 31;
  const int ntiles = Nqkv / 32;            // 96
  int mt = wave / ntiles, nt = wave - mt * ntiles;
  int m0 = mt * 32, n0 = nt * 32;

  v8f acc00 = {}, acc01 = {}, acc10 = {}, acc11 = {};
  for (int k0 = 0; k0 < Cdim; k0 += 32) {
    __builtin_prefetch((const void*)(Xb + (size_t)m0 * Cdim + k0 + 64), 0, 1);
    v16bf a0 = load_a_frag(Xb, Cdim, m0,      k0, lane);
    v16bf a1 = load_a_frag(Xb, Cdim, m0 + 16, k0, lane);
    v16bf b0 = load_b_frag(Wt, Nqkv, k0, n0,      lane);
    v16bf b1 = load_b_frag(Wt, Nqkv, k0, n0 + 16, lane);
    acc00 = wmma_bf16(a0, b0, acc00);
    acc01 = wmma_bf16(a0, b1, acc01);
    acc10 = wmma_bf16(a1, b0, acc10);
    acc11 = wmma_bf16(a1, b1, acc11);
  }

  int rowoff = (lane >> 4) << 3;           // 0 or 8
  int cl = lane & 15;
  v8f* accs[4] = { &acc00, &acc01, &acc10, &acc11 };
  for (int ij = 0; ij < 4; ++ij) {
    int i = ij >> 1, j = ij & 1;
    v8f a = *accs[ij];
    for (int r = 0; r < 8; ++r) {
      int row = m0 + i * 16 + rowoff + r;  // b*T + t
      int col = n0 + j * 16 + cl;          // 0..3071
      int which = col >> 10;               // 0=q 1=k 2=v
      int c = col & (Cdim - 1);
      int h = c >> 6, d = c & (HS - 1);
      int b = row >> 11, t = row & (Tdim - 1);
      size_t bh = (size_t)(b * Hn + h);
      bf16 bv = (bf16)a[r];
      if (which == 0)      Q [(bh * Tdim + t) * HS + d]  = bv;
      else if (which == 1) Kt[(bh * HS + d) * Tdim + t]  = bv;   // transposed
      else                 V [(bh * Tdim + t) * HS + d]  = bv;
    }
  }
}

// ---------------------------------------------------------------------------
// Flash-style sp2norm attention, block-cooperative:
//   block = 8 waves = 128 queries of one (b,h); K/V chunks (32 keys) are
//   staged once per block into double-buffered LDS via async global->LDS
//   copies (ASYNCcnt) and consumed by all 8 waves through strided per-element LDS reads.
//   Per chunk/wave: S = Q*Kt (2 WMMA k-steps x 2 halves), softplus, causal
//   mask, sum(w^2) accum, register-transpose of W through a wave-private LDS
//   tile, then O += W @ V (4 WMMA).
__global__ __launch_bounds__(256) void attn_kernel(const bf16* __restrict__ Q,
                                                   const bf16* __restrict__ Kt,
                                                   const bf16* __restrict__ V,
                                                   bf16* __restrict__ Att) {
  __shared__ __align__(16) bf16 Ksh[2][64 * 32];  // [d][key] per chunk, 4KB each
  __shared__ __align__(16) bf16 Vsh[2][32 * 64];  // [key][d] per chunk, 4KB each
  __shared__ __align__(16) bf16 Wsh[8][16 * 32];  // per-wave W transpose tiles

  int tid   = threadIdx.x;
  int wslot = tid >> 5;
  int lane  = tid & 31;

  int bh   = blockIdx.x >> 4;              // 16 query-blocks per (b,h)
  int qblk = blockIdx.x & 15;
  int qb0  = qblk * 128;
  int q0   = qb0 + wslot * 16;             // this wave's query tile

  const bf16* Qp  = Q  + (size_t)bh * Tdim * HS;
  const bf16* Ktp = Kt + (size_t)bh * HS * Tdim;
  const bf16* Vp  = V  + (size_t)bh * Tdim * HS;

  // Q tile (16 x 64) as two A fragments (K=32 each)
  v16bf qa0 = load_a_frag(Qp, HS, q0, 0,  lane);
  v16bf qa1 = load_a_frag(Qp, HS, q0, 32, lane);

  bf16* wt = Wsh[wslot];
  v8f o0 = {}, o1 = {}, o2 = {}, o3 = {};
  float nsq[8] = {0.f, 0.f, 0.f, 0.f, 0.f, 0.f, 0.f, 0.f};

  int m  = lane & 15;
  int kb = (lane >> 4) << 3;
  int rowoff = kb;                          // C-layout row split
  int cl = lane & 15;
  const float scale = 0.125f;               // 1/sqrt(64)

  // stage helper (256 threads x 16B each for K and for V => 4KB + 4KB)
  auto stage = [&](int buf, int k0) {
    {
      int d = tid >> 2, seg = (tid & 3) << 3;           // K chunk: [64][32]
      copy16(&Ksh[buf][d * 32 + seg], Ktp + (size_t)d * Tdim + k0 + seg);
    }
    {
      int key = tid >> 3, seg = (tid & 7) << 3;         // V chunk: [32][64]
      copy16(&Vsh[buf][key * 64 + seg], Vp + (size_t)(k0 + key) * HS + seg);
    }
  };

  int nchunks = (qb0 >> 5) + 4;            // cover block diagonal
  stage(0, 0);
  for (int j = 0; j < nchunks; ++j) {
    int cur = j & 1;
    int k0  = j << 5;
    if (j + 1 < nchunks) {
      stage(cur ^ 1, k0 + 32);
      wait_async_pending();                // chunk j landed; j+1 in flight
    } else {
      wait_async_all();
    }
    __syncthreads();                        // chunk j visible block-wide

    if (k0 <= q0 + 15) {                    // causally live for this wave
      const bf16* Kb = Ksh[cur];
      const bf16* Vb = Vsh[cur];
      v8f s0 = {}, s1 = {};
      s0 = wmma_bf16(qa0, load_b_frag(Kb, 32, 0,  0,  lane), s0);
      s0 = wmma_bf16(qa1, load_b_frag(Kb, 32, 32, 0,  lane), s0);
      s1 = wmma_bf16(qa0, load_b_frag(Kb, 32, 0,  16, lane), s1);
      s1 = wmma_bf16(qa1, load_b_frag(Kb, 32, 32, 16, lane), s1);

      for (int r = 0; r < 8; ++r) {
        int qrow = q0 + rowoff + r;
        float w0 = softplus_f(s0[r] * scale);
        if (k0 + cl > qrow) w0 = 0.0f;
        float w1 = softplus_f(s1[r] * scale);
        if (k0 + 16 + cl > qrow) w1 = 0.0f;
        nsq[r] += w0 * w0 + w1 * w1;
        wt[(rowoff + r) * 32 + cl]      = (bf16)w0;
        wt[(rowoff + r) * 32 + cl + 16] = (bf16)w1;
      }

      // reload W chunk as 16x32 A fragment (wave-private tile, dscnt-ordered)
      ABFrag wf;
      wf.h[0] = *(const bf16x8*)&wt[m * 32 + kb];
      wf.h[1] = *(const bf16x8*)&wt[m * 32 + 16 + kb];

      o0 = wmma_bf16(wf.v, load_b_frag(Vb, 64, 0, 0,  lane), o0);
      o1 = wmma_bf16(wf.v, load_b_frag(Vb, 64, 0, 16, lane), o1);
      o2 = wmma_bf16(wf.v, load_b_frag(Vb, 64, 0, 32, lane), o2);
      o3 = wmma_bf16(wf.v, load_b_frag(Vb, 64, 0, 48, lane), o3);
    }
    __syncthreads();                        // buffer may be overwritten next
  }

  // reduce sum(w^2) across the 16-lane half-group (matches C-layout rows)
  float rinv[8];
  for (int r = 0; r < 8; ++r) {
    float s = nsq[r];
    s += __shfl_xor(s, 1, 32);
    s += __shfl_xor(s, 2, 32);
    s += __shfl_xor(s, 4, 32);
    s += __shfl_xor(s, 8, 32);
    rinv[r] = __frsqrt_rn(s);
  }

  // store normalized O into Att [B,T,C] (bf16)
  int b = bh >> 4, h = bh & (Hn - 1);
  v8f* os[4] = { &o0, &o1, &o2, &o3 };
  for (int n0 = 0; n0 < 4; ++n0) {
    v8f ov = *os[n0];
    for (int r = 0; r < 8; ++r) {
      size_t t = (size_t)q0 + rowoff + r;
      Att[((size_t)b * Tdim + t) * Cdim + h * HS + n0 * 16 + cl] =
          (bf16)(ov[r] * rinv[r]);
    }
  }
}

// ---------------------------------------------------------------------------
// Output projection: Y = Att @ Wproj^T + bias, fp32 out. 32x32 tile per wave.
__global__ __launch_bounds__(256) void proj_gemm_kernel(const bf16* __restrict__ Att,
                                                        const bf16* __restrict__ Wpt,
                                                        const float* __restrict__ bias,
                                                        float* __restrict__ Y) {
  int wave = blockIdx.x * (blockDim.x >> 5) + (threadIdx.x >> 5);
  int lane = threadIdx.x & 31;
  const int ntiles = Cdim / 32;            // 32
  int mt = wave / ntiles, nt = wave - mt * ntiles;
  int m0 = mt * 32, n0 = nt * 32;

  v8f acc00 = {}, acc01 = {}, acc10 = {}, acc11 = {};
  for (int k0 = 0; k0 < Cdim; k0 += 32) {
    __builtin_prefetch((const void*)(Att + (size_t)m0 * Cdim + k0 + 64), 0, 1);
    v16bf a0 = load_a_frag(Att, Cdim, m0,      k0, lane);
    v16bf a1 = load_a_frag(Att, Cdim, m0 + 16, k0, lane);
    v16bf b0 = load_b_frag(Wpt, Cdim, k0, n0,      lane);
    v16bf b1 = load_b_frag(Wpt, Cdim, k0, n0 + 16, lane);
    acc00 = wmma_bf16(a0, b0, acc00);
    acc01 = wmma_bf16(a0, b1, acc01);
    acc10 = wmma_bf16(a1, b0, acc10);
    acc11 = wmma_bf16(a1, b1, acc11);
  }

  int rowoff = (lane >> 4) << 3;
  int cl = lane & 15;
  v8f* accs[4] = { &acc00, &acc01, &acc10, &acc11 };
  for (int ij = 0; ij < 4; ++ij) {
    int i = ij >> 1, j = ij & 1;
    v8f a = *accs[ij];
    for (int r = 0; r < 8; ++r) {
      int row = m0 + i * 16 + rowoff + r;
      int col = n0 + j * 16 + cl;
      Y[(size_t)row * Cdim + col] = a[r] + bias[col];
    }
  }
}

// ---------------------------------------------------------------------------
extern "C" void kernel_launch(void* const* d_in, const int* in_sizes, int n_in,
                              void* d_out, int out_size, void* d_ws, size_t ws_size,
                              hipStream_t stream) {
  const float* x      = (const float*)d_in[0];   // [B,T,C]
  const float* W_qkv  = (const float*)d_in[1];   // [3C,C]
  const float* W_proj = (const float*)d_in[2];   // [C,C]
  const float* b_proj = (const float*)d_in[3];   // [C]
  float* y = (float*)d_out;                      // [B,T,C]

  char* ws = (char*)d_ws;
  bf16* Xb     = (bf16*)(ws);                                   //  8 MiB
  bf16* Wqkvt  = (bf16*)(ws + (8u  << 20));                     //  6 MiB
  bf16* Wprojt = (bf16*)(ws + (14u << 20));                     //  2 MiB
  bf16* Qb     = (bf16*)(ws + (16u << 20));                     //  8 MiB
  bf16* Ktb    = (bf16*)(ws + (24u << 20));                     //  8 MiB
  bf16* Vb     = (bf16*)(ws + (32u << 20));                     //  8 MiB
  bf16* Attb   = (bf16*)(ws + (40u << 20));                     //  8 MiB (48 total)

  // 1) converts / transposes
  cvt_bf16_kernel<<<(Mrows * Cdim) / 256, 256, 0, stream>>>(x, Xb, Mrows * Cdim);
  cvt_t_bf16_kernel<<<(Nqkv * Cdim) / 256, 256, 0, stream>>>(W_qkv, Wqkvt, Nqkv, Cdim);
  cvt_t_bf16_kernel<<<(Cdim * Cdim) / 256, 256, 0, stream>>>(W_proj, Wprojt, Cdim, Cdim);

  // 2) QKV projection GEMM -> Q / Kt / V
  {
    int waves = (Mrows / 32) * (Nqkv / 32);   // 12288
    qkv_gemm_kernel<<<waves / 8, 256, 0, stream>>>(Xb, Wqkvt, Qb, Ktb, Vb);
  }

  // 3) fused sp2norm flash attention (block-cooperative, async LDS staging)
  {
    int blocks = Bn * Hn * (Tdim / 128);      // 512
    attn_kernel<<<blocks, 256, 0, stream>>>(Qb, Ktb, Vb, Attb);
  }

  // 4) output projection + bias
  {
    int waves = (Mrows / 32) * (Cdim / 32);   // 4096
    proj_gemm_kernel<<<waves / 8, 256, 0, stream>>>(Attb, Wprojt, b_proj, y);
  }
}
